// Block_54924041781446
// MI455X (gfx1250) — matrix operand order repaired and checked by reading, not compiled, not run on hardware
//
#include <hip/hip_runtime.h>
#include <hip/hip_bf16.h>
#include <math.h>

#define EMB 1024
#define HEADS 16
#define HD 64
#define HID 4096
#define BATCH 2
#define SEQ 2048
#define MTOK (BATCH * SEQ)

typedef __attribute__((ext_vector_type(16))) __bf16 bf16x16;
typedef __attribute__((ext_vector_type(8)))  __bf16 bf16x8;
typedef __attribute__((ext_vector_type(8)))  float  v8f;

static __device__ __forceinline__ v8f v8f_zero() {
  v8f z;
#pragma unroll
  for (int i = 0; i < 8; ++i) z[i] = 0.f;
  return z;
}

static __device__ __forceinline__ v8f wmma_bf16(bf16x16 a, bf16x16 b, v8f c) {
  // D = A(16x32 bf16) * B(32x16 bf16) + C(16x16 f32)
  return __builtin_amdgcn_wmma_f32_16x16x32_bf16(false, a, false, b, (short)0, c,
                                                 false, false);
}

// ---------------------------------------------------------------- f32 -> bf16
__global__ __launch_bounds__(256) void f2bf_kernel(const float* __restrict__ src,
                                                   __bf16* __restrict__ dst, int n) {
  int i = blockIdx.x * 256 + threadIdx.x;
  if (i < n) dst[i] = (__bf16)src[i];
}

// ---------------------------------------------------------------- LayerNorm
// One block (256 thr = 8 waves) per token. Output normalized bf16.
__global__ __launch_bounds__(256) void layernorm_bf16(const float* __restrict__ x,
                                                      const float* __restrict__ w,
                                                      const float* __restrict__ b,
                                                      __bf16* __restrict__ out) {
  int t = blockIdx.x;
  const float* xr = x + (size_t)t * EMB;
  int lane = threadIdx.x & 31, wid = threadIdx.x >> 5;
  float s = 0.f, ss = 0.f;
  for (int i = threadIdx.x; i < EMB; i += 256) {
    float v = xr[i];
    s += v;
    ss += v * v;
  }
#pragma unroll
  for (int off = 16; off >= 1; off >>= 1) {
    s += __shfl_down(s, off, 32);
    ss += __shfl_down(ss, off, 32);
  }
  __shared__ float sh_s[8], sh_ss[8];
  if (lane == 0) { sh_s[wid] = s; sh_ss[wid] = ss; }
  __syncthreads();
  float ts = 0.f, tss = 0.f;
#pragma unroll
  for (int j = 0; j < 8; ++j) { ts += sh_s[j]; tss += sh_ss[j]; }
  float mu = ts * (1.f / EMB);
  float var = tss * (1.f / EMB) - mu * mu;
  float rs = rsqrtf(var + 1e-6f);
  for (int i = threadIdx.x; i < EMB; i += 256) {
    out[(size_t)t * EMB + i] = (__bf16)((xr[i] - mu) * rs * w[i] + b[i]);
  }
}

// ---------------------------------------------------------------- WMMA GEMM
// C[M,N] = A[M,K] * W[N,K]^T + bias, fused epilogue.
// EPI: 1 = QKV scatter -> bf16 [3][B,H,S,D]
//      2 = GELU(exact) -> bf16 [M,N]
//      3 = +resid      -> f32  [M,N]
// Block: 256 thr = 8 waves; block tile 128(M) x 128(N); wave tile 32 x 64.
// Inner loop is register double-buffered: k+1 fragments are in flight while
// k's eight WMMAs execute, hiding L2 latency behind the matrix pipe.
template <int EPI>
__global__ __launch_bounds__(256) void gemm_bf16(const __bf16* __restrict__ A,
                                                 const __bf16* __restrict__ W,
                                                 const float* __restrict__ bias,
                                                 const float* __restrict__ resid,
                                                 void* __restrict__ outp,
                                                 int M, int N, int K) {
  int lane = threadIdx.x & 31, wid = threadIdx.x >> 5;
  int half = lane >> 4, lm = lane & 15;
  int wm = wid & 3, wn = wid >> 2;
  int m0 = blockIdx.y * 128 + wm * 32;
  int n0 = blockIdx.x * 128 + wn * 64;
  const __bf16* arow0 = A + (size_t)(m0 + lm) * K;
  const __bf16* arow1 = arow0 + (size_t)16 * K;
  const __bf16* wbase = W + (size_t)(n0 + lm) * K + half * 16;

  v8f acc[2][4];
#pragma unroll
  for (int i = 0; i < 2; ++i)
#pragma unroll
    for (int t = 0; t < 4; ++t) acc[i][t] = v8f_zero();

  // A fragment 16x32: lanes 0-15 hold K[0..15], lanes 16-31 K[16..31]
  auto loadA = [&](const __bf16* row, int k0) {
    bf16x8 lo = *(const bf16x8*)(row + k0 + half * 16);
    bf16x8 hi = *(const bf16x8*)(row + k0 + half * 16 + 8);
    bf16x16 r;
#pragma unroll
    for (int j = 0; j < 8; ++j) { r[j] = lo[j]; r[8 + j] = hi[j]; }
    return r;
  };

  bf16x16 a0c = loadA(arow0, 0);
  bf16x16 a1c = loadA(arow1, 0);
  bf16x16 bc[4];
#pragma unroll
  for (int t = 0; t < 4; ++t)
    bc[t] = *(const bf16x16*)(wbase + (size_t)t * 16 * K);

  for (int k0 = 0; k0 < K; k0 += 32) {
    int kn = k0 + 32;
    bf16x16 a0n, a1n, bn[4];
    if (kn < K) {
      a0n = loadA(arow0, kn);
      a1n = loadA(arow1, kn);
#pragma unroll
      for (int t = 0; t < 4; ++t)
        bn[t] = *(const bf16x16*)(wbase + (size_t)t * 16 * K + kn);
      __builtin_prefetch((const void*)(arow0 + kn + 32), 0, 0);
    } else {
      a0n = a0c; a1n = a1c;
#pragma unroll
      for (int t = 0; t < 4; ++t) bn[t] = bc[t];
    }
#pragma unroll
    for (int t = 0; t < 4; ++t) {
      acc[0][t] = wmma_bf16(a0c, bc[t], acc[0][t]);
      acc[1][t] = wmma_bf16(a1c, bc[t], acc[1][t]);
    }
    a0c = a0n; a1c = a1n;
#pragma unroll
    for (int t = 0; t < 4; ++t) bc[t] = bn[t];
  }

#pragma unroll
  for (int i = 0; i < 2; ++i) {
#pragma unroll
    for (int t = 0; t < 4; ++t) {
#pragma unroll
      for (int r = 0; r < 8; ++r) {
        int m = m0 + i * 16 + r + 8 * half;  // C layout: VGPR r, lane-half -> M
        int n = n0 + t * 16 + lm;
        float v = acc[i][t][r] + bias[n];
        if (EPI == 1) {
          // scatter to q/k/v bf16 [part][B,H,S,D] (part/head constant per subtile)
          int part = n >> 10;
          int e = n & (EMB - 1);
          int h = e >> 6;
          int d = e & (HD - 1);
          int b = m >> 11;           // / SEQ
          int sdx = m & (SEQ - 1);   // % SEQ
          size_t dst = (size_t)part * ((size_t)MTOK * EMB) +
                       ((size_t)(b * HEADS + h) * SEQ + sdx) * HD + d;
          ((__bf16*)outp)[dst] = (__bf16)v;
        } else if (EPI == 2) {
          v = 0.5f * v * (1.f + erff(v * 0.70710678f));
          ((__bf16*)outp)[(size_t)m * N + n] = (__bf16)v;
        } else {
          size_t idx = (size_t)m * N + n;
          ((float*)outp)[idx] = v + resid[idx];
        }
      }
    }
  }
}

// ---------------------------------------------------------------- Attention
// Flash-style causal attention. Block = 128 thr = 4 waves; each wave owns a
// 16-row query tile and streams key tiles of 32 with online softmax.
// Per 32 keys: 4 score WMMAs + 4 dense PV WMMAs. K fragments are register
// double-buffered so loads for tile kt+1 overlap tile kt's WMMA + softmax.
// grid = (B*H, SEQ/64)
__global__ __launch_bounds__(128) void attention_kernel(const __bf16* __restrict__ q,
                                                        const __bf16* __restrict__ k,
                                                        const __bf16* __restrict__ v,
                                                        __bf16* __restrict__ out) {
  __shared__ float lds_p[4][16 * 32];  // per-wave P staging (C-layout -> A-layout)
  int lane = threadIdx.x & 31, wid = threadIdx.x >> 5;
  int half = lane >> 4, lm = lane & 15;
  int bh = blockIdx.x;
  int b = bh >> 4, h = bh & 15;
  int qt = blockIdx.y * 4 + wid;
  int q0 = qt * 16;
  const __bf16* qb = q + ((size_t)bh * SEQ + q0) * HD;
  const __bf16* kb = k + (size_t)bh * SEQ * HD;
  const __bf16* vb = v + (size_t)bh * SEQ * HD;
  float* myp = lds_p[wid];

  // Q fragments for the two K=32 chunks of D=64
  bf16x16 qf[2];
#pragma unroll
  for (int c = 0; c < 2; ++c) {
    const __bf16* p = qb + (size_t)lm * HD + c * 32;
    bf16x8 lo = *(const bf16x8*)(p + half * 16);
    bf16x8 hi = *(const bf16x8*)(p + half * 16 + 8);
#pragma unroll
    for (int j = 0; j < 8; ++j) { qf[c][j] = lo[j]; qf[c][8 + j] = hi[j]; }
  }

  v8f acc[4];
#pragma unroll
  for (int t = 0; t < 4; ++t) acc[t] = v8f_zero();
  float mi[8], li[8];
#pragma unroll
  for (int r = 0; r < 8; ++r) { mi[r] = -__builtin_inff(); li[r] = 0.f; }

  // K-tile fragments: [key subtile st][d chunk c]
  auto loadK = [&](int k0, bf16x16 kf[2][2]) {
#pragma unroll
    for (int st = 0; st < 2; ++st)
#pragma unroll
      for (int c = 0; c < 2; ++c)
        kf[st][c] = *(const bf16x16*)(kb + (size_t)(k0 + st * 16 + lm) * HD +
                                      c * 32 + half * 16);
  };

  int nkt = (q0 + 15) / 32 + 1;  // 32-key tiles needed for causal range
  bf16x16 kfc[2][2];
  loadK(0, kfc);

  for (int kt = 0; kt < nkt; ++kt) {
    int k0 = kt * 32;
    bf16x16 kfn[2][2];
    if (kt + 1 < nkt) {
      loadK(k0 + 32, kfn);
    } else {
#pragma unroll
      for (int st = 0; st < 2; ++st)
#pragma unroll
        for (int c = 0; c < 2; ++c) kfn[st][c] = kfc[st][c];
    }

    // scores 16x32 = Q(16x64) . K_tile(32x64)^T, scale 1/sqrt(64)
    v8f sc[2];
#pragma unroll
    for (int st = 0; st < 2; ++st)
      sc[st] = wmma_bf16(qf[1], kfc[st][1], wmma_bf16(qf[0], kfc[st][0], v8f_zero()));
#pragma unroll
    for (int st = 0; st < 2; ++st)
#pragma unroll
      for (int r = 0; r < 8; ++r) sc[st][r] *= 0.125f;
    if (k0 + 31 > q0) {  // tile crosses the causal diagonal
#pragma unroll
      for (int st = 0; st < 2; ++st)
#pragma unroll
        for (int r = 0; r < 8; ++r) {
          int i = q0 + r + 8 * half;
          int j = k0 + st * 16 + lm;
          if (j > i) sc[st][r] = -__builtin_inff();
        }
    }

    // online softmax row statistics (rows live across 16 lanes of a half-wave)
    float pr0[8], pr1[8], alpha[8];
#pragma unroll
    for (int r = 0; r < 8; ++r) {
      float mx = fmaxf(sc[0][r], sc[1][r]);
#pragma unroll
      for (int msk = 1; msk <= 8; msk <<= 1) mx = fmaxf(mx, __shfl_xor(mx, msk, 32));
      float nm = fmaxf(mi[r], mx);
      alpha[r] = __expf(mi[r] - nm);
      float p0 = __expf(sc[0][r] - nm);
      float p1 = __expf(sc[1][r] - nm);
      float rs = p0 + p1;
#pragma unroll
      for (int msk = 1; msk <= 8; msk <<= 1) rs += __shfl_xor(rs, msk, 32);
      li[r] = li[r] * alpha[r] + rs;
      mi[r] = nm;
      pr0[r] = p0;
      pr1[r] = p1;
    }
#pragma unroll
    for (int t = 0; t < 4; ++t)
#pragma unroll
      for (int r = 0; r < 8; ++r) acc[t][r] *= alpha[r];

    // P tile 16x32: C-layout -> A-layout via per-wave LDS
#pragma unroll
    for (int r = 0; r < 8; ++r) {
      myp[(r + 8 * half) * 32 + lm] = pr0[r];
      myp[(r + 8 * half) * 32 + 16 + lm] = pr1[r];
    }
    __asm__ volatile("s_wait_dscnt 0x0" ::: "memory");
    bf16x16 pa;  // dense 16x32 A fragment
#pragma unroll
    for (int j = 0; j < 8; ++j) {
      pa[j] = (__bf16)myp[lm * 32 + half * 16 + j];
      pa[8 + j] = (__bf16)myp[lm * 32 + half * 16 + 8 + j];
    }

    // acc += P(16x32) . V(32x16): all 32 lanes carry V rows (K = half*16 + idx)
    bf16x16 vf[4];
#pragma unroll
    for (int t = 0; t < 4; ++t)
#pragma unroll
      for (int idx = 0; idx < 16; ++idx)
        vf[t][idx] = vb[(size_t)(k0 + half * 16 + idx) * HD + t * 16 + lm];
#pragma unroll
    for (int t = 0; t < 4; ++t) acc[t] = wmma_bf16(pa, vf[t], acc[t]);

#pragma unroll
    for (int st = 0; st < 2; ++st)
#pragma unroll
      for (int c = 0; c < 2; ++c) kfc[st][c] = kfn[st][c];
  }

  float ri[8];
#pragma unroll
  for (int r = 0; r < 8; ++r) ri[r] = 1.f / li[r];
#pragma unroll
  for (int t = 0; t < 4; ++t)
#pragma unroll
    for (int r = 0; r < 8; ++r) {
      int m = q0 + r + 8 * half;
      int n = h * HD + t * 16 + lm;
      out[(size_t)(b * SEQ + m) * EMB + n] = (__bf16)(acc[t][r] * ri[r]);
    }
}

// ---------------------------------------------------------------- launch
extern "C" void kernel_launch(void* const* d_in, const int* in_sizes, int n_in,
                              void* d_out, int out_size, void* d_ws, size_t ws_size,
                              hipStream_t stream) {
  const float* x      = (const float*)d_in[0];
  const float* ln1_w  = (const float*)d_in[1];
  const float* ln1_b  = (const float*)d_in[2];
  const float* qkv_w  = (const float*)d_in[3];
  const float* qkv_b  = (const float*)d_in[4];
  const float* proj_w = (const float*)d_in[5];
  const float* proj_b = (const float*)d_in[6];
  const float* ln2_w  = (const float*)d_in[7];
  const float* ln2_b  = (const float*)d_in[8];
  const float* fc1_w  = (const float*)d_in[9];
  const float* fc1_b  = (const float*)d_in[10];
  const float* fc2_w  = (const float*)d_in[11];
  const float* fc2_b  = (const float*)d_in[12];
  float* out = (float*)d_out;

  // workspace carving (256B aligned)
  char* p = (char*)d_ws;
  auto alloc = [&](size_t bytes) {
    char* r = p;
    p += (bytes + 255) & ~(size_t)255;
    return r;
  };
  __bf16* wqkv_bf  = (__bf16*)alloc((size_t)3 * EMB * EMB * 2);
  __bf16* wproj_bf = (__bf16*)alloc((size_t)EMB * EMB * 2);
  __bf16* wfc1_bf  = (__bf16*)alloc((size_t)HID * EMB * 2);
  __bf16* wfc2_bf  = (__bf16*)alloc((size_t)EMB * HID * 2);
  __bf16* ln1_bf   = (__bf16*)alloc((size_t)MTOK * EMB * 2);
  __bf16* ln2_bf   = (__bf16*)alloc((size_t)MTOK * EMB * 2);
  __bf16* qkv_bf   = (__bf16*)alloc((size_t)3 * MTOK * EMB * 2);  // q|k|v [B,H,S,D]
  __bf16* attn_bf  = (__bf16*)alloc((size_t)MTOK * EMB * 2);
  float*  x1_f     = (float*)alloc((size_t)MTOK * EMB * 4);
  __bf16* h_bf     = (__bf16*)alloc((size_t)MTOK * HID * 2);
  __bf16* q_bf = qkv_bf;
  __bf16* k_bf = qkv_bf + (size_t)MTOK * EMB;
  __bf16* v_bf = qkv_bf + (size_t)2 * MTOK * EMB;
  (void)ws_size; (void)n_in; (void)in_sizes; (void)out_size;

  // 1) weights -> bf16
  {
    int n = 3 * EMB * EMB;
    f2bf_kernel<<<(n + 255) / 256, 256, 0, stream>>>(qkv_w, wqkv_bf, n);
    n = EMB * EMB;
    f2bf_kernel<<<(n + 255) / 256, 256, 0, stream>>>(proj_w, wproj_bf, n);
    n = HID * EMB;
    f2bf_kernel<<<(n + 255) / 256, 256, 0, stream>>>(fc1_w, wfc1_bf, n);
    n = EMB * HID;
    f2bf_kernel<<<(n + 255) / 256, 256, 0, stream>>>(fc2_w, wfc2_bf, n);
  }
  // 2) LN1
  layernorm_bf16<<<MTOK, 256, 0, stream>>>(x, ln1_w, ln1_b, ln1_bf);
  // 3) QKV GEMM with fused bf16 [B,H,S,D] scatter
  gemm_bf16<1><<<dim3(3 * EMB / 128, MTOK / 128), 256, 0, stream>>>(
      ln1_bf, wqkv_bf, qkv_b, nullptr, qkv_bf, MTOK, 3 * EMB, EMB);
  // 4) causal flash attention -> [M,E] bf16
  attention_kernel<<<dim3(BATCH * HEADS, SEQ / 64), 128, 0, stream>>>(q_bf, k_bf,
                                                                      v_bf, attn_bf);
  // 5) proj GEMM + bias + residual(x) -> x1 f32
  gemm_bf16<3><<<dim3(EMB / 128, MTOK / 128), 256, 0, stream>>>(
      attn_bf, wproj_bf, proj_b, x, x1_f, MTOK, EMB, EMB);
  // 6) LN2
  layernorm_bf16<<<MTOK, 256, 0, stream>>>(x1_f, ln2_w, ln2_b, ln2_bf);
  // 7) FC1 GEMM + bias + GELU -> bf16
  gemm_bf16<2><<<dim3(HID / 128, MTOK / 128), 256, 0, stream>>>(
      ln2_bf, wfc1_bf, fc1_b, nullptr, h_bf, MTOK, HID, EMB);
  // 8) FC2 GEMM + bias + residual(x1) -> d_out f32
  gemm_bf16<3><<<dim3(EMB / 128, MTOK / 128), 256, 0, stream>>>(
      h_bf, wfc2_bf, fc2_b, x1_f, out, MTOK, EMB, HID);
}
